// CrissCrossAttention_71073118814729
// MI455X (gfx1250) — matrix operand order repaired and checked by reading, not compiled, not run on hardware
//
#include <hip/hip_runtime.h>

// CDNA5 / gfx1250: wave32, WMMA bf16 16x16x32, f32 accumulate.
typedef __attribute__((ext_vector_type(16))) __bf16 v16bf;
typedef __attribute__((ext_vector_type(8)))  __bf16 v8bf;
typedef __attribute__((ext_vector_type(8)))  float  v8f;

static constexpr int B  = 16, C = 512, V = 96, T = 96, C8 = 64;
static constexpr int N  = V * T;                       // 9216
// workspace (bf16 elements):
//   q2  [B][N][C8]  (c contiguous)   : SQ
//   k2  [B][N][C8]                   : SQ
//   val [B][C][N]   (t contiguous)   : SVAL
//   val2[B][T][C][V] (v contiguous)  : SVAL
//   aH  [B][V][T][V] (u contiguous)  : SA
//   aW  [B][V][T][T] (s contiguous)  : SA
static constexpr size_t SQ   = (size_t)B * C8 * N;     //  9,437,184
static constexpr size_t SVAL = (size_t)B * C  * N;     // 75,497,472
static constexpr size_t SA   = (size_t)B * V * T * V;  // 14,155,776
// total = 2*SQ + 2*SVAL + 2*SA = 198,180,864 bf16 = ~378 MB of d_ws

// wave-uniform flat wave id, forced into an SGPR so tile decode / branches /
// base addresses are scalar.
__device__ __forceinline__ int wave_id() {
  return __builtin_amdgcn_readfirstlane((int)((blockIdx.x * blockDim.x + threadIdx.x) >> 5));
}

// --- fragment loaders for K-contiguous memory (ISA VGPR layouts) -----------
// A (16x32 bf16): lane half h: element e -> K = (e<8 ? e : e+8) + 8h
__device__ __forceinline__ v16bf frag_a(const __bf16* p, int k0, int half) {
  const v8bf lo = *(const v8bf*)(p + k0 + 8 * half);
  const v8bf hi = *(const v8bf*)(p + k0 + 8 * half + 16);
  return __builtin_shufflevector(lo, hi, 0,1,2,3,4,5,6,7,8,9,10,11,12,13,14,15);
}
// B (32x16 bf16): lane half h: element e -> K = (e<8 ? e : e+8) + 8h
__device__ __forceinline__ v16bf frag_b(const __bf16* p, int k0, int half) {
  const v8bf lo = *(const v8bf*)(p + k0 + 8 * half);
  const v8bf hi = *(const v8bf*)(p + k0 + 8 * half + 16);
  return __builtin_shufflevector(lo, hi, 0,1,2,3,4,5,6,7,8,9,10,11,12,13,14,15);
}

// ---------------------------------------------------------------------------
// Stage 1: fused projection GEMM [Wq;Wk;Wv](640x512) x x_b(512x9216).
// ---------------------------------------------------------------------------
__global__ __launch_bounds__(256) void cc_proj(
    const float* __restrict__ x,
    const float* __restrict__ Wq, const float* __restrict__ bq,
    const float* __restrict__ Wk, const float* __restrict__ bk,
    const float* __restrict__ Wv, const float* __restrict__ bv,
    __bf16* __restrict__ q2, __bf16* __restrict__ k2, __bf16* __restrict__ vo)
{
  const int wid  = wave_id();
  const int lane = threadIdx.x & 31;
  const int half = lane >> 4, l15 = lane & 15;

  const int ntile = wid % (N / 16);
  int t2 = wid / (N / 16);
  const int mtile = t2 % 40;
  const int b     = t2 / 40;
  if (b >= B) return;                       // scalar branch

  const int m0 = mtile * 16, n0 = ntile * 16;

  // scalar region selection (tiles never straddle 64/128 boundaries)
  const float* Wbase; const float* biasp; int mrel0;
  if (m0 < C8)          { Wbase = Wq; biasp = bq; mrel0 = m0;          }
  else if (m0 < 2 * C8) { Wbase = Wk; biasp = bk; mrel0 = m0 - C8;     }
  else                  { Wbase = Wv; biasp = bv; mrel0 = m0 - 2 * C8; }

  const float* wp = Wbase + (size_t)(mrel0 + l15) * C + 8 * half;
  const float* px = x + (size_t)b * C * N + (size_t)(8 * half) * N + n0 + l15;

  v8f acc = {};
  for (int k0 = 0; k0 < C; k0 += 32) {
    const float4 w0 = *(const float4*)(wp);
    const float4 w1 = *(const float4*)(wp + 4);
    const float4 w2 = *(const float4*)(wp + 16);
    const float4 w3 = *(const float4*)(wp + 20);
    v16bf a;
    a[0]=(__bf16)w0.x; a[1]=(__bf16)w0.y; a[2]=(__bf16)w0.z; a[3]=(__bf16)w0.w;
    a[4]=(__bf16)w1.x; a[5]=(__bf16)w1.y; a[6]=(__bf16)w1.z; a[7]=(__bf16)w1.w;
    a[8]=(__bf16)w2.x; a[9]=(__bf16)w2.y; a[10]=(__bf16)w2.z; a[11]=(__bf16)w2.w;
    a[12]=(__bf16)w3.x; a[13]=(__bf16)w3.y; a[14]=(__bf16)w3.z; a[15]=(__bf16)w3.w;
    v16bf bm;
#pragma unroll
    for (int e = 0; e < 16; ++e)
      bm[e] = (__bf16)px[(size_t)(e < 8 ? e : e + 8) * N];    // e*N folds into the ioffset
    acc = __builtin_amdgcn_wmma_f32_16x16x32_bf16(false, a, false, bm,
                                                  (short)0, acc, false, false);
    wp += 32;
    px += (size_t)32 * N;
  }

  const int n = n0 + l15;
  const float* bp = biasp + mrel0 + 8 * half;
  if (m0 < 2 * C8) {
    __bf16* outqk = (m0 < C8) ? q2 : k2;   // scalar select
    v8bf st;
#pragma unroll
    for (int r = 0; r < 8; ++r)
      st[r] = (__bf16)(acc[r] + bp[r]);
    *(v8bf*)(outqk + ((size_t)b * N + n) * C8 + mrel0 + 8 * half) = st;
  } else {
    __bf16* pv = vo + ((size_t)b * C + mrel0 + 8 * half) * N + n;
#pragma unroll
    for (int r = 0; r < 8; ++r)
      pv[(size_t)r * N] = (__bf16)(acc[r] + bp[r]);
  }
}

// ---------------------------------------------------------------------------
// Stage 1b: val2[b][t][c][v] = val[b][c][v*T+t]  (LDS-tiled 32x32 transpose).
// Inbound leg uses CDNA5 async global->LDS copies (ASYNCcnt) when available:
// no VGPR staging, fill overlaps the write-out address setup.
// ---------------------------------------------------------------------------
__global__ __launch_bounds__(256) void cc_tr_val(const __bf16* __restrict__ val,
                                                 __bf16* __restrict__ val2)
{
  __shared__ __bf16 lds[32][34];           // row stride 68 B: 4B-aligned, odd dwords
  int tile = blockIdx.x;
  const int tv = tile % 3; tile /= 3;      // v tile
  const int tt = tile % 3; tile /= 3;      // t tile
  const int c  = tile % C;
  const int b  = tile / C;
  const int v0 = tv * 32, t0 = tt * 32;

  const int tx = threadIdx.x & 31, ty = threadIdx.x >> 5;   // ty: 0..7
  const __bf16* src = val + ((size_t)b * C + c) * N;

#if __has_builtin(__builtin_amdgcn_global_load_async_to_lds_b32)
  // 32x32 bf16 tile = 512 dwords; 2 async B32 transfers per thread.
#pragma unroll
  for (int i = 0; i < 2; ++i) {
    const int w   = i * 256 + (int)threadIdx.x;   // 0..511
    const int row = w >> 4;                       // v row 0..31
    const int col = (w & 15) * 2;                 // t column pair (4B aligned)
    __builtin_amdgcn_global_load_async_to_lds_b32(
        (__attribute__((address_space(1))) int*)
            (src + (size_t)(v0 + row) * T + t0 + col),
        (__attribute__((address_space(3))) int*)&lds[row][col],
        0, 0);
  }
#if __has_builtin(__builtin_amdgcn_s_wait_asynccnt)
  __builtin_amdgcn_s_wait_asynccnt(0);
#else
  asm volatile("s_wait_asynccnt 0x0" ::: "memory");
#endif
#else
  // fallback: VGPR-staged fill
#pragma unroll
  for (int i = 0; i < 32; i += 8)
    lds[ty + i][tx] = src[(size_t)(v0 + ty + i) * T + t0 + tx];
#endif
  __syncthreads();

  __bf16* dst = val2 + ((size_t)(b * T + t0 + ty) * C + c) * V + v0 + tx;
#pragma unroll
  for (int i = 0; i < 32; i += 8)
    dst[(size_t)i * C * V] = lds[tx][ty + i];               // coalesced in v
}

// ---------------------------------------------------------------------------
// Stage 2a: eH[b,v,t,u] = sum_c q.k ; diagonal masked.
// ---------------------------------------------------------------------------
__global__ __launch_bounds__(256) void cc_scores_h(
    const __bf16* __restrict__ q2, const __bf16* __restrict__ k2,
    __bf16* __restrict__ aH)
{
  const int wid  = wave_id();
  const int lane = threadIdx.x & 31;
  const int half = lane >> 4, l15 = lane & 15;

  const int ut = wid % 6;
  int t2 = wid / 6;
  const int vt = t2 % 6; t2 /= 6;
  const int t  = t2 % T;
  const int b  = t2 / T;
  if (b >= B) return;

  const int v0 = vt * 16, u0 = ut * 16;
  const __bf16* pa = q2 + ((size_t)b * N + (v0 + l15) * T + t) * C8;
  const __bf16* pb = k2 + ((size_t)b * N + (u0 + l15) * T + t) * C8;

  v8f acc = {};
#pragma unroll
  for (int c0 = 0; c0 < C8; c0 += 32)
    acc = __builtin_amdgcn_wmma_f32_16x16x32_bf16(false, frag_a(pa, c0, half),
                                                  false, frag_b(pb, c0, half),
                                                  (short)0, acc, false, false);

  const int u = u0 + l15;
  __bf16* ph = aH + (((size_t)b * V + v0 + 8 * half) * T + t) * V + u;
#pragma unroll
  for (int r = 0; r < 8; ++r) {
    const int v = v0 + r + half * 8;
    const float s = (v == u) ? -1e30f : acc[r];
    ph[(size_t)r * T * V] = (__bf16)s;                 // coalesced in u
  }
}

// ---------------------------------------------------------------------------
// Stage 2b: eW[b,v,t,s] = sum_c q.k.
// ---------------------------------------------------------------------------
__global__ __launch_bounds__(256) void cc_scores_w(
    const __bf16* __restrict__ q2, const __bf16* __restrict__ k2,
    __bf16* __restrict__ aW)
{
  const int wid  = wave_id();
  const int lane = threadIdx.x & 31;
  const int half = lane >> 4, l15 = lane & 15;

  const int st = wid % 6;
  int t2 = wid / 6;
  const int tt = t2 % 6; t2 /= 6;
  const int v  = t2 % V;
  const int b  = t2 / V;
  if (b >= B) return;

  const int t0 = tt * 16, s0 = st * 16;
  const __bf16* pa = q2 + ((size_t)b * N + v * T + (t0 + l15)) * C8;
  const __bf16* pb = k2 + ((size_t)b * N + v * T + (s0 + l15)) * C8;

  v8f acc = {};
#pragma unroll
  for (int c0 = 0; c0 < C8; c0 += 32)
    acc = __builtin_amdgcn_wmma_f32_16x16x32_bf16(false, frag_a(pa, c0, half),
                                                  false, frag_b(pb, c0, half),
                                                  (short)0, acc, false, false);

  __bf16* pw = aW + ((size_t)(b * V + v) * T + t0 + 8 * half) * T + s0 + l15;
#pragma unroll
  for (int r = 0; r < 8; ++r)
    pw[(size_t)r * T] = (__bf16)acc[r];                // coalesced in s
}

// ---------------------------------------------------------------------------
// Stage 3: softmax over concat([eH row, eW row]) per (b,v,t). One wave/row.
// ---------------------------------------------------------------------------
__global__ __launch_bounds__(256) void cc_softmax(__bf16* __restrict__ aH,
                                                  __bf16* __restrict__ aW)
{
  const int row  = wave_id();
  const int lane = threadIdx.x & 31;
  if (row >= B * V * T) return;

  __bf16* h = aH + (size_t)row * V + lane * 3;
  __bf16* w = aW + (size_t)row * T + lane * 3;

  float hv[3], wv[3];
  float m = -3.0e38f;
#pragma unroll
  for (int i = 0; i < 3; ++i) {
    hv[i] = (float)h[i];
    wv[i] = (float)w[i];
    m = fmaxf(m, fmaxf(hv[i], wv[i]));
  }
#pragma unroll
  for (int d = 16; d >= 1; d >>= 1) m = fmaxf(m, __shfl_xor(m, d, 32));

  float s = 0.f;
#pragma unroll
  for (int i = 0; i < 3; ++i) {
    hv[i] = __expf(hv[i] - m); s += hv[i];
    wv[i] = __expf(wv[i] - m); s += wv[i];
  }
#pragma unroll
  for (int d = 16; d >= 1; d >>= 1) s += __shfl_xor(s, d, 32);
  const float inv = 1.0f / s;

#pragma unroll
  for (int i = 0; i < 3; ++i) {
    h[i] = (__bf16)(hv[i] * inv);
    w[i] = (__bf16)(wv[i] * inv);
  }
}

// ---------------------------------------------------------------------------
// Stage 4a: outH[b,c,v,t] = sum_u val2[b,t,c,u] aH[b,v,t,u]; d_out = g*outH.
// ---------------------------------------------------------------------------
__global__ __launch_bounds__(256) void cc_out_h(
    const __bf16* __restrict__ val2, const __bf16* __restrict__ aH,
    const float* __restrict__ gamma, float* __restrict__ out)
{
  const int wid  = wave_id();
  const int lane = threadIdx.x & 31;
  const int half = lane >> 4, l15 = lane & 15;

  const int t  = wid % T;            // fastest: 8 consecutive t per block
  int t2 = wid / T;
  const int vt = t2 % 6; t2 /= 6;
  const int ct = t2 % 32;
  const int b  = t2 / 32;
  if (b >= B) return;

  const int c0 = ct * 16, v0 = vt * 16;
  const __bf16* pa = val2 + ((size_t)(b * T + t) * C + c0 + l15) * V;      // u-contig
  const __bf16* pb = aH   + (((size_t)b * V + v0 + l15) * T + t) * V;      // u-contig

  v8f acc = {};
#pragma unroll
  for (int k0 = 0; k0 < V; k0 += 32)
    acc = __builtin_amdgcn_wmma_f32_16x16x32_bf16(false, frag_a(pa, k0, half),
                                                  false, frag_b(pb, k0, half),
                                                  (short)0, acc, false, false);

  const float g = gamma[0];
  float* po = out + (((size_t)b * C + c0 + 8 * half) * V + v0 + l15) * T + t;
#pragma unroll
  for (int r = 0; r < 8; ++r)
    po[(size_t)r * V * T] = g * acc[r];
}

// ---------------------------------------------------------------------------
// Stage 4b: outW[b,c,v,t] = sum_s val[b,c,v,s] aW[b,v,t,s];
// d_out += g*outW + x (disjoint tiles; stream order makes 4a visible).
// ---------------------------------------------------------------------------
__global__ __launch_bounds__(256) void cc_out_w(
    const __bf16* __restrict__ val, const __bf16* __restrict__ aW,
    const float* __restrict__ gamma, const float* __restrict__ x,
    float* __restrict__ out)
{
  const int wid  = wave_id();
  const int lane = threadIdx.x & 31;
  const int half = lane >> 4, l15 = lane & 15;

  const int tt = wid % 6;
  int t2 = wid / 6;
  const int ct = t2 % 32; t2 /= 32;
  const int v  = t2 % V;
  const int b  = t2 / V;
  if (b >= B) return;

  const int c0 = ct * 16, t0 = tt * 16;
  const __bf16* pa = val + ((size_t)b * C + c0 + l15) * N + v * T;         // s-contig
  const __bf16* pb = aW  + ((size_t)(b * V + v) * T + t0 + l15) * T;       // s-contig

  v8f acc = {};
#pragma unroll
  for (int k0 = 0; k0 < T; k0 += 32)
    acc = __builtin_amdgcn_wmma_f32_16x16x32_bf16(false, frag_a(pa, k0, half),
                                                  false, frag_b(pb, k0, half),
                                                  (short)0, acc, false, false);

  const float g = gamma[0];
  const size_t obase = (((size_t)b * C + c0 + 8 * half) * V + v) * T + t0 + l15;
  float* po = out + obase;
  const float* pxr = x + obase;
#pragma unroll
  for (int r = 0; r < 8; ++r)
    po[(size_t)r * V * T] = po[(size_t)r * V * T] + g * acc[r] + pxr[(size_t)r * V * T];
}

// ---------------------------------------------------------------------------
extern "C" void kernel_launch(void* const* d_in, const int* in_sizes, int n_in,
                              void* d_out, int out_size, void* d_ws, size_t ws_size,
                              hipStream_t stream)
{
  (void)in_sizes; (void)n_in; (void)out_size; (void)ws_size;
  const float* x     = (const float*)d_in[0];
  const float* Wq    = (const float*)d_in[1];
  const float* bq    = (const float*)d_in[2];
  const float* Wk    = (const float*)d_in[3];
  const float* bk    = (const float*)d_in[4];
  const float* Wv    = (const float*)d_in[5];
  const float* bv    = (const float*)d_in[6];
  const float* gamma = (const float*)d_in[7];
  float* out = (float*)d_out;

  __bf16* ws   = (__bf16*)d_ws;            // ~378 MB used
  __bf16* q2   = ws;
  __bf16* k2   = ws + SQ;
  __bf16* val  = ws + 2 * SQ;
  __bf16* val2 = val + SVAL;
  __bf16* aH   = val2 + SVAL;
  __bf16* aW   = aH + SA;

  const dim3 blk(256);                     // 8 waves/block
  cc_proj    <<<46080, blk, 0, stream>>>(x, Wq, bq, Wk, bk, Wv, bv, q2, k2, val);
  cc_tr_val  <<<73728, blk, 0, stream>>>(val, val2);
  cc_scores_h<<< 6912, blk, 0, stream>>>(q2, k2, aH);
  cc_scores_w<<< 6912, blk, 0, stream>>>(q2, k2, aW);
  cc_softmax <<<18432, blk, 0, stream>>>(aH, aW);
  cc_out_h   <<<36864, blk, 0, stream>>>(val2, aH, gamma, out);
  cc_out_w   <<<36864, blk, 0, stream>>>(val, aW, gamma, x, out);
}
